// StructureEncoder_87995289960561
// MI455X (gfx1250) — matrix operand order repaired and checked by reading, not compiled, not run on hardware
//
#include <hip/hip_runtime.h>
#include <hip/hip_bf16.h>

// ---------------------------------------------------------------------------
// StructureEncoder for MI455X (gfx1250): bf16 WMMA, one wave32 per 16-row tile
// 3-slot liveness-scheduled parent-feature buffer -> 19.2KB LDS/wave
// ---------------------------------------------------------------------------
typedef __attribute__((ext_vector_type(16))) __bf16 v16bf;
typedef __attribute__((ext_vector_type(8)))  float  v8f;

#define KJ 24
#define FEAT 64
#define NIN 77           // 13 + 64
#define KPAD 96          // padded K for both per-joint GEMMs (3 x 32)
#define TILE 16          // batch rows per wave (WMMA M)
#define FRAGH 512        // halfs per B-fragment (32 lanes x 16 halfs)
#define NSLOT 3          // max simultaneously-live parent features (tree liveness)

// PARENTS = [-1,0,0,0,1,2,3,4,5,6,7,8,9,9,9,12,13,14,16,17,18,19,20,21]
// Greedy slot schedule (slot freed after last child's x-build):
//   PSLOT[i] = slot holding parent feature when joint i runs (prior -> slot 0)
//   OSLOT[i] = slot to store joint i's feature into (-1 for leaves 10,11,15,22,23)
__device__ __constant__ int PSLOT[KJ] = {
  0, 0, 0, 0, 1, 2, 0, 1, 2, 0, 1, 2, 0, 0, 0, 1, 2, 0, 1, 2, 0, 1, 2, 0
};
__device__ __constant__ int OSLOT[KJ] = {
  0, 1, 2, 0, 1, 2, 0, 1, 2, 0, -1, -1, 1, 2, 0, -1, 1, 2, 0, 1, 2, 0, -1, -1
};

// ws fragment layout (units of FRAGH halfs):
//   [0,36)            : Wp   frags, idx = kc*4 + nt   (kc<9,  nt<4)
//   [36, 36+24*15)    : W1[i] frags, idx = kc*5 + nt  (kc<3,  nt<5)
//   [396, 396+24*12)  : W2[i] frags, idx = kc*4 + nt  (kc<3,  nt<4)
#define WS_W1 36
#define WS_W2 (36 + KJ*15)
#define NFRAG (36 + KJ*15 + KJ*12)   // 684

__device__ inline v8f wmma_bf16(v16bf a, v16bf b, v8f c) {
  return __builtin_amdgcn_wmma_f32_16x16x32_bf16(false, a, false, b, (short)0, c,
                                                 false, false);
}

__device__ inline unsigned pack2(float a, float b) {
  union { __bf16 h[2]; unsigned u; } t;
  t.h[0] = (__bf16)a; t.h[1] = (__bf16)b;
  return t.u;
}

// A-fragment (16x32 bf16) from an LDS row vector; contiguous k order (same as B packing):
//   lane<16 holds K = kc*32 + {0..15}; lane>=16 holds {16..31}
__device__ inline v16bf lds_loadA(const __bf16* rowp, int kc, int half) {
  const uint4* q = (const uint4*)(rowp + kc * 32 + half * 16);
  union { uint4 q2[2]; v16bf v; } u;
  u.q2[0] = q[0];   // K base+0..7   (16 bytes)
  u.q2[1] = q[1];   // K base+8..15  (16 bytes)
  return u.v;
}

// B-fragment: packed per lane, 32 contiguous bytes (default RT hint: L2-resident)
__device__ inline v16bf load_B(const __bf16* frag, int lane) {
  const uint4* q = (const uint4*)(frag + lane * 16);
  union { uint4 q2[2]; v16bf v; } u;
  u.q2[0] = q[0];
  u.q2[1] = q[1];
  return u.v;
}

// ---------------------------------------------------------------------------
// Pack f32 weights -> bf16 WMMA B-fragments in ws (one block per fragment)
// B layout (32x16 bf16): lane<16 holds K=0..15 (t=K), lane>=16 holds K=16..31
// ---------------------------------------------------------------------------
__global__ __launch_bounds__(32)
void pack_weights(const float* __restrict__ Wp, const float* __restrict__ W1,
                  const float* __restrict__ W2, __bf16* __restrict__ ws) {
  const int fid  = blockIdx.x;
  const int lane = threadIdx.x;
  const int n    = lane & 15;
  const int kbase = (lane >> 4) * 16;

  int mode, joint = 0, kc, nt;
  if (fid < 36) {
    mode = 0; kc = fid / 4; nt = fid % 4;
  } else if (fid < WS_W2) {
    int r = fid - WS_W1; mode = 1; joint = r / 15; r %= 15; kc = r / 5; nt = r % 5;
  } else {
    int r = fid - WS_W2; mode = 2; joint = r / 12; r %= 12; kc = r / 4; nt = r % 4;
  }

  union { __bf16 h[16]; uint4 q[2]; } u;
#pragma unroll
  for (int t = 0; t < 16; ++t) {
    const int K = kc * 32 + kbase + t;
    const int N = nt * 16 + n;
    float v = 0.f;
    if (mode == 0) {                       // Wp: 288 x 64, no padding needed
      if (K < 288 && N < 64) v = Wp[K * 64 + N];
    } else if (mode == 1) {                // W1: padded-x remap (x[13] is pad)
      int sk = (K < 13) ? K : ((K >= 14 && K < 78) ? K - 1 : -1);
      if (sk >= 0 && N < NIN) v = W1[(joint * NIN + sk) * NIN + N];
    } else {                               // W2: plain h layout
      if (K < NIN && N < FEAT) v = W2[(joint * NIN + K) * FEAT + N];
    }
    u.h[t] = (__bf16)v;
  }
  uint4* dst = (uint4*)(ws + (size_t)fid * FRAGH + lane * 16);
  dst[0] = u.q[0];
  dst[1] = u.q[1];
}

// ---------------------------------------------------------------------------
// Main kernel: one wave32 per 16-row batch tile; sequential walk over joints.
// Dynamic LDS (wave-private; single-wave workgroup => no barriers, LDS in-order):
//   bf288 : [16][288] bf16  bone features (Wp GEMM A source)        9216 B
//   blen  : [16][24]  bf16  bone lengths                             768 B
//   xbuf  : [16][96]  bf16  x / h staging (padded K)                3072 B
//   feats : [3][16][64] bf16  liveness-scheduled parent features    6144 B
// total 19200 B -> ~16 workgroups (waves) per WGP
// ---------------------------------------------------------------------------
__global__ __launch_bounds__(32)
void se_main(const float* __restrict__ pose, const float* __restrict__ rel,
             const float* __restrict__ bp,   const float* __restrict__ b1,
             const float* __restrict__ b2,   const __bf16* __restrict__ ws,
             float* __restrict__ out) {
  extern __shared__ __align__(16) char smem[];
  __bf16* bf288 = (__bf16*)smem;                 // 16*288
  __bf16* blen  = bf288 + TILE * 288;            // 16*24
  __bf16* xbuf  = blen  + TILE * KJ;             // 16*96
  __bf16* feats = xbuf  + TILE * KPAD;           // 3*16*64

  const int lane = threadIdx.x;
  const int row  = lane & 15;     // A-operand M index / D-operand N index
  const int half = lane >> 4;
  const int r0   = blockIdx.x * TILE;
  const int b    = r0 + row;

  // ---- Stage inputs (streamed once -> non-temporal loads) ----------------
  // lanes split the 24 joints of their row: half0 -> j 0..11, half1 -> 12..23
#pragma unroll
  for (int jj = 0; jj < 12; ++jj) {
    const int j = half * 12 + jj;
    const float* pp = pose + (size_t)b * (KJ * 9) + (size_t)j * 9;
    const float* rr = rel  + (size_t)b * (KJ * 3) + (size_t)j * 3;
    float p0 = __builtin_nontemporal_load(pp + 0);
    float p1 = __builtin_nontemporal_load(pp + 1);
    float p2 = __builtin_nontemporal_load(pp + 2);
    float p3 = __builtin_nontemporal_load(pp + 3);
    float p4 = __builtin_nontemporal_load(pp + 4);
    float p5 = __builtin_nontemporal_load(pp + 5);
    float p6 = __builtin_nontemporal_load(pp + 6);
    float p7 = __builtin_nontemporal_load(pp + 7);
    float p8 = __builtin_nontemporal_load(pp + 8);
    float q0 = __builtin_nontemporal_load(rr + 0);
    float q1 = __builtin_nontemporal_load(rr + 1);
    float q2 = __builtin_nontemporal_load(rr + 2);
    const float len = sqrtf(q0 * q0 + q1 * q1 + q2 * q2);
    unsigned* drow = (unsigned*)(bf288 + row * 288 + j * 12);  // 4B aligned
    drow[0] = pack2(p0, p1); drow[1] = pack2(p2, p3); drow[2] = pack2(p4, p5);
    drow[3] = pack2(p6, p7); drow[4] = pack2(p8, q0); drow[5] = pack2(q1, q2);
    blen[row * KJ + j] = (__bf16)len;
  }

  // ---- Root prior: (16x288) @ (288x64) + bp, NO relu -> feats slot 0 -----
  {
    v8f acc[4];
#pragma unroll
    for (int nt = 0; nt < 4; ++nt) {
      const float bv = bp[nt * 16 + row];
#pragma unroll
      for (int g = 0; g < 8; ++g) acc[nt][g] = bv;
    }
    const __bf16* arow = bf288 + row * 288;
#pragma unroll
    for (int kc = 0; kc < 9; ++kc) {
      const v16bf a = lds_loadA(arow, kc, half);
#pragma unroll
      for (int nt = 0; nt < 4; ++nt) {
        const v16bf bm = load_B(ws + (size_t)(kc * 4 + nt) * FRAGH, lane);
        acc[nt] = wmma_bf16(a, bm, acc[nt]);
      }
    }
    // D layout: n = lane&15, m = g + 8*half
#pragma unroll
    for (int nt = 0; nt < 4; ++nt)
#pragma unroll
      for (int g = 0; g < 8; ++g) {
        const int m = g + half * 8;
        feats[(0 * TILE + m) * FEAT + nt * 16 + row] = (__bf16)acc[nt][g];
      }
  }

  // ---- Kinematic-tree walk ----------------------------------------------
  const __bf16* w1base = ws + (size_t)WS_W1 * FRAGH;
  const __bf16* w2base = ws + (size_t)WS_W2 * FRAGH;

  for (int i = 0; i < KJ; ++i) {
    const int pslot = PSLOT[i];
    const int oslot = OSLOT[i];

    // Build x = [bone13 | pad | parent_feat64 | pad] (padded layout, K=96).
    // All copies dword-aligned; wave-private LDS keeps DS ops in order.
    {
      const unsigned* fsrc = (const unsigned*)(feats + (pslot * TILE + row) * FEAT);
      unsigned* xd = (unsigned*)(xbuf + row * KPAD);
      if (half == 0) {
        const unsigned* bsrc = (const unsigned*)(bf288 + row * 288 + i * 12);
#pragma unroll
        for (int d = 0; d < 6; ++d) xd[d] = bsrc[d];       // halfs 0..11
        xd[6] = pack2((float)blen[row * KJ + i], 0.f);     // halfs 12,13 (pad)
#pragma unroll
        for (int d = 0; d < 17; ++d) xd[7 + d] = fsrc[d];  // halfs 14..47
      } else {
#pragma unroll
        for (int d = 0; d < 15; ++d) xd[24 + d] = fsrc[17 + d]; // halfs 48..77
#pragma unroll
        for (int d = 0; d < 9; ++d) xd[39 + d] = 0u;            // halfs 78..95
      }
    }

    const __bf16* xrow = xbuf + row * KPAD;

    // GEMM1: h = relu(x @ W1[i] + b1[i])   (16x96)@(96x80) -> 16x80
    v8f h[5];
#pragma unroll
    for (int nt = 0; nt < 5; ++nt) {
      const int N = nt * 16 + row;
      const float bv = (N < NIN) ? b1[i * NIN + N] : 0.f;
#pragma unroll
      for (int g = 0; g < 8; ++g) h[nt][g] = bv;
    }
    const __bf16* w1 = w1base + (size_t)i * 15 * FRAGH;
#pragma unroll
    for (int kc = 0; kc < 3; ++kc) {
      const v16bf a = lds_loadA(xrow, kc, half);
#pragma unroll
      for (int nt = 0; nt < 5; ++nt) {
        const v16bf bm = load_B(w1 + (size_t)(kc * 5 + nt) * FRAGH, lane);
        h[nt] = wmma_bf16(a, bm, h[nt]);
      }
    }
    // relu(h) -> xbuf plain layout (cols 77..79 are exact zeros; 80..95 stay 0)
#pragma unroll
    for (int nt = 0; nt < 5; ++nt)
#pragma unroll
      for (int g = 0; g < 8; ++g) {
        const int m = g + half * 8;
        xbuf[m * KPAD + nt * 16 + row] = (__bf16)fmaxf(h[nt][g], 0.f);
      }

    // GEMM2: f = relu(h @ W2[i] + b2[i])   (16x96)@(96x64) -> 16x64
    v8f f[4];
#pragma unroll
    for (int nt = 0; nt < 4; ++nt) {
      const float bv = b2[i * FEAT + nt * 16 + row];
#pragma unroll
      for (int g = 0; g < 8; ++g) f[nt][g] = bv;
    }
    const __bf16* w2 = w2base + (size_t)i * 12 * FRAGH;
#pragma unroll
    for (int kc = 0; kc < 3; ++kc) {
      const v16bf a = lds_loadA(xrow, kc, half);
#pragma unroll
      for (int nt = 0; nt < 4; ++nt) {
        const v16bf bm = load_B(w2 + (size_t)(kc * 4 + nt) * FRAGH, lane);
        f[nt] = wmma_bf16(a, bm, f[nt]);
      }
    }
    // relu -> feats slot (bf16, children only) + NT f32 output (64B runs)
#pragma unroll
    for (int nt = 0; nt < 4; ++nt)
#pragma unroll
      for (int g = 0; g < 8; ++g) {
        const int m = g + half * 8;
        const int n = nt * 16 + row;
        const float v = fmaxf(f[nt][g], 0.f);
        if (oslot >= 0)   // uniform (scalar) branch; leaves skip the LDS store
          feats[(oslot * TILE + m) * FEAT + n] = (__bf16)v;
        __builtin_nontemporal_store(
            v, out + (size_t)(r0 + m) * (KJ * FEAT) + i * FEAT + n);
      }
  }
}

// ---------------------------------------------------------------------------
extern "C" void kernel_launch(void* const* d_in, const int* in_sizes, int n_in,
                              void* d_out, int out_size, void* d_ws, size_t ws_size,
                              hipStream_t stream) {
  const float* pose = (const float*)d_in[0];
  const float* rel  = (const float*)d_in[1];
  const float* Wp   = (const float*)d_in[2];
  const float* bp   = (const float*)d_in[3];
  const float* W1   = (const float*)d_in[4];
  const float* b1   = (const float*)d_in[5];
  const float* W2   = (const float*)d_in[6];
  const float* b2   = (const float*)d_in[7];
  float* out = (float*)d_out;
  __bf16* ws = (__bf16*)d_ws;

  const int Bn = in_sizes[1] / (KJ * 3);   // batch size (65536)

  pack_weights<<<NFRAG, 32, 0, stream>>>(Wp, W1, W2, ws);

  const size_t lds_bytes =
      (size_t)(TILE * 288 + TILE * KJ + TILE * KPAD + NSLOT * TILE * FEAT) * 2;
  se_main<<<Bn / TILE, 32, lds_bytes, stream>>>(pose, rel, bp, b1, b2, ws, out);
}
